// MoeLayer_68693706932328
// MI455X (gfx1250) — matrix operand order repaired and checked by reading, not compiled, not run on hardware
//
#include <hip/hip_runtime.h>

#define T_TOK 2048
#define D_MOD 1024
#define H_FFN 3584
#define N_EXP 8

typedef __attribute__((ext_vector_type(16))) __bf16 v16bf;
typedef __attribute__((ext_vector_type(8)))  float  v8f;
typedef __attribute__((ext_vector_type(4)))  unsigned int v4u;
typedef __attribute__((ext_vector_type(8)))  int v8i_;
typedef __attribute__((ext_vector_type(4)))  int v4i_;

union Frag {
    v16bf v;
    uint4 q[2];
};

__device__ __forceinline__ unsigned short f2bf(float f) {
    unsigned int u = __float_as_uint(f);
    u += 0x7FFFu + ((u >> 16) & 1u);   // round-to-nearest-even
    return (unsigned short)(u >> 16);
}

// ---------------------------------------------------------------------------
// fp32 -> bf16 conversion (grid-stride)
// ---------------------------------------------------------------------------
__global__ void cvt_bf16(const float* __restrict__ src,
                         unsigned short* __restrict__ dst, long n) {
    long i = (long)blockIdx.x * blockDim.x + threadIdx.x;
    long stride = (long)gridDim.x * blockDim.x;
    for (; i < n; i += stride) dst[i] = f2bf(src[i]);
}

// ---------------------------------------------------------------------------
// Zero-init output (needed because expert halves combine via fp32 atomics)
// ---------------------------------------------------------------------------
__global__ void zero_f32(float* __restrict__ dst, long n) {
    long i = (long)blockIdx.x * blockDim.x + threadIdx.x;
    long stride = (long)gridDim.x * blockDim.x;
    for (; i < n; i += stride) dst[i] = 0.0f;
}

// ---------------------------------------------------------------------------
// Gate: logits [T,8], top-2 with lower-index tie-break, softmax over the 2,
// scatter into dense comb [T,8]
// ---------------------------------------------------------------------------
__global__ __launch_bounds__(256) void gate_topk(const float* __restrict__ x,
                                                 const float* __restrict__ Wg,
                                                 float* __restrict__ comb) {
    __shared__ float wgs[N_EXP * D_MOD];   // 32 KB
    const int t = blockIdx.x * 256 + threadIdx.x;
    for (int i = threadIdx.x; i < N_EXP * D_MOD; i += 256) wgs[i] = Wg[i];
    __syncthreads();

    float acc[N_EXP];
#pragma unroll
    for (int e = 0; e < N_EXP; ++e) acc[e] = 0.0f;

    const float* xr = x + (size_t)t * D_MOD;
    for (int d = 0; d < D_MOD; ++d) {
        float xv = xr[d];
#pragma unroll
        for (int e = 0; e < N_EXP; ++e) acc[e] += xv * wgs[e * D_MOD + d];
    }

    int   i0 = 0;
    float v0 = acc[0];
#pragma unroll
    for (int e = 1; e < N_EXP; ++e)
        if (acc[e] > v0) { v0 = acc[e]; i0 = e; }
    int   i1 = -1;
    float v1 = -3.4e38f;
#pragma unroll
    for (int e = 0; e < N_EXP; ++e)
        if (e != i0 && acc[e] > v1) { v1 = acc[e]; i1 = e; }

    float e1 = __expf(v1 - v0);          // v0 >= v1, numerically safe
    float s  = 1.0f + e1;
    float w0 = 1.0f / s;
    float w1 = e1 / s;

    float* cr = comb + (size_t)t * N_EXP;
#pragma unroll
    for (int e = 0; e < N_EXP; ++e) cr[e] = 0.0f;
    cr[i0] = w0;
    cr[i1] = w1;
}

// ---------------------------------------------------------------------------
// Main fused MoE kernel.
// grid = (T/16, 2): block = 16 tokens x 4 experts (blockIdx.y picks the
// expert half). 256 threads = 8 waves. Per expert: GEMM1 (h,g over H in
// 128-wide chunks, each wave one 16x16 h-tile + g-tile, K over D=1024),
// silu*g -> LDS (double-buffered), GEMM2 (each wave owns 128 of the D=1024
// output columns), comb-scaled accumulation. The two expert halves combine
// into d_out via fp32 atomics (exactly 2 adds/element -> commutative ->
// bit-deterministic).
// ---------------------------------------------------------------------------
__global__ __launch_bounds__(256) void moe_main(
    const unsigned short* __restrict__ xb,
    const unsigned short* __restrict__ w1b,
    const unsigned short* __restrict__ w3b,
    const unsigned short* __restrict__ w2b,
    const float* __restrict__ comb,
    float* __restrict__ out) {
    __shared__ unsigned short xs[16 * D_MOD];      // 32 KB x tile (bf16)
    __shared__ unsigned short acts[2][16 * 128];   // 8 KB double-buffered act
    __shared__ float combs[16 * N_EXP];            // combine weights for tile

    const int tile = blockIdx.x;
    const int e0   = blockIdx.y * (N_EXP / 2);     // expert half
    const int tid  = threadIdx.x;
    const int wv   = tid >> 5;          // wave id 0..7
    const int lane = tid & 31;
    const int hl   = lane & 15;         // half-lane index
    const int grp  = lane >> 4;         // 0 = lanes 0-15, 1 = lanes 16-31

    const unsigned short* xg = xb + (size_t)tile * 16 * D_MOD;

#if __has_builtin(__builtin_amdgcn_tensor_load_to_lds)
    // Tensor Data Mover: one contiguous 32KB global->LDS copy, issued once
    // by wave 0; D# = {count=1, lds_addr, global_addr, type=2} +
    // {data_size=2B, tensor 16384x1, tile 16384x1, stride 16384}.
    if (wv == 0) {
        unsigned lds_off = (unsigned)(uintptr_t)(&xs[0]);
        unsigned long long ga = (unsigned long long)(uintptr_t)xg;
        v4u g0;
        g0[0] = 1u;                                   // count=1 (valid D#)
        g0[1] = lds_off;                              // lds_addr
        g0[2] = (unsigned)(ga & 0xFFFFFFFFu);         // global_addr[31:0]
        g0[3] = (unsigned)((ga >> 32) & 0x1FFFFFFu)   // global_addr[56:32]
                | (2u << 30);                         // type=2 ("image")
        v8i_ g1;
        g1[0] = 0x00010000;          // wg_mask=0, data_size=1 (2 bytes)
        g1[1] = 0x40000000;          // tensor_dim0[15:0]=16384 @ [63:48]
        g1[2] = 0x00010000;          // tensor_dim0 hi=0, tensor_dim1=1
        g1[3] = 0x40000000;          // tile_dim0=16384 @ [127:112]
        g1[4] = 1;                   // tile_dim1=1, tile_dim2=0
        g1[5] = 16384;               // tensor_dim0_stride[31:0]
        g1[6] = 0x40000000;          // stride hi=0, tensor_dim1_stride lo
        g1[7] = 0;
        v4i_ z4 = {0, 0, 0, 0};
#if defined(__clang_major__) && __clang_major__ >= 23
        v8i_ z8 = {0, 0, 0, 0, 0, 0, 0, 0};
        __builtin_amdgcn_tensor_load_to_lds(g0, g1, z4, z4, z8, 0);
#else
        __builtin_amdgcn_tensor_load_to_lds(g0, g1, z4, z4, 0);
#endif
        __builtin_amdgcn_s_wait_tensorcnt(0);
    }
#else
    for (int i = tid * 8; i < 16 * D_MOD; i += 256 * 8)
        *(uint4*)&xs[i] = *(const uint4*)&xg[i];
#endif
    if (tid < 16 * N_EXP) combs[tid] = comb[(size_t)tile * 16 * N_EXP + tid];
    __syncthreads();

    float tot[8][8];
#pragma unroll
    for (int nt = 0; nt < 8; ++nt)
#pragma unroll
        for (int i = 0; i < 8; ++i) tot[nt][i] = 0.0f;

    for (int e = e0; e < e0 + N_EXP / 2; ++e) {
        v8f c2[8];
#pragma unroll
        for (int nt = 0; nt < 8; ++nt) c2[nt] = (v8f){0, 0, 0, 0, 0, 0, 0, 0};

        for (int hc = 0; hc < H_FFN / 128; ++hc) {
            const int h0 = hc * 128 + wv * 16;   // this wave's 16 H columns
            unsigned short* ab = acts[hc & 1];   // double buffer

            v8f ch = (v8f){0, 0, 0, 0, 0, 0, 0, 0};
            v8f cg = (v8f){0, 0, 0, 0, 0, 0, 0, 0};

            // B rows: w1[e][h0+hl][*], w3[e][h0+hl][*]; K split by lane group.
            const unsigned short* b1p =
                w1b + ((size_t)(e * H_FFN + h0 + hl)) * D_MOD + grp * 16;
            const unsigned short* b3p =
                w3b + ((size_t)(e * H_FFN + h0 + hl)) * D_MOD + grp * 16;
            const unsigned short* ap0 = &xs[hl * D_MOD + grp * 16];

#pragma unroll 4
            for (int kb = 0; kb < D_MOD / 32; ++kb) {
                Frag A;
                const unsigned short* ap = ap0 + kb * 32;
                A.q[0] = *(const uint4*)(ap);        // K = kb*32+g*16 .. +7
                A.q[1] = *(const uint4*)(ap + 8);    // K = kb*32+g*16+8 .. +15
                Frag B1, B3;
                B1.q[0] = *(const uint4*)(b1p + kb * 32);
                B1.q[1] = *(const uint4*)(b1p + kb * 32 + 8);
                B3.q[0] = *(const uint4*)(b3p + kb * 32);
                B3.q[1] = *(const uint4*)(b3p + kb * 32 + 8);
                ch = __builtin_amdgcn_wmma_f32_16x16x32_bf16(
                    false, A.v, false, B1.v, (short)0, ch, false, false);
                cg = __builtin_amdgcn_wmma_f32_16x16x32_bf16(
                    false, A.v, false, B3.v, (short)0, cg, false, false);
            }

            // silu(h) * g -> bf16 activation chunk in LDS.
            // v_rcp_f32 instead of the IEEE division chain: the whole
            // pipeline is bf16-quantized, so ~1ulp rcp is the right cost
            // model and these VALU/TRANS ops co-execute with the XDL WMMAs.
#pragma unroll
            for (int i = 0; i < 8; ++i) {
                float hv  = ch[i];
                float gv  = cg[i];
                float sig = __builtin_amdgcn_rcpf(1.0f + __expf(-hv));
                float a   = hv * sig * gv;
                int row   = grp ? (i + 8) : i;
                ab[row * 128 + wv * 16 + hl] = f2bf(a);
            }
            __syncthreads();

            // GEMM2: this wave owns D columns [wv*128, wv*128+128)
#pragma unroll
            for (int kk = 0; kk < 4; ++kk) {
                Frag A;
                const unsigned short* ap = &ab[hl * 128 + kk * 32 + grp * 16];
                A.q[0] = *(const uint4*)(ap);
                A.q[1] = *(const uint4*)(ap + 8);
#pragma unroll
                for (int nt = 0; nt < 8; ++nt) {
                    const int d0 = wv * 128 + nt * 16;
                    const unsigned short* bp =
                        w2b + ((size_t)(e * D_MOD + d0 + hl)) * H_FFN +
                        hc * 128 + kk * 32 + grp * 16;
                    Frag B;
                    B.q[0] = *(const uint4*)(bp);
                    B.q[1] = *(const uint4*)(bp + 8);
                    c2[nt] = __builtin_amdgcn_wmma_f32_16x16x32_bf16(
                        false, A.v, false, B.v, (short)0, c2[nt], false, false);
                }
            }
            // no second barrier: next chunk writes the other acts buffer
        }

        // comb-scaled accumulation into totals
        float cwv[8];
#pragma unroll
        for (int i = 0; i < 8; ++i)
            cwv[i] = combs[(grp ? (i + 8) : i) * N_EXP + e];
#pragma unroll
        for (int nt = 0; nt < 8; ++nt)
#pragma unroll
            for (int i = 0; i < 8; ++i) tot[nt][i] += cwv[i] * c2[nt][i];
    }

    // combine the two expert halves: exactly 2 fp32 adds per element
    // (commutative -> bit-deterministic)
#pragma unroll
    for (int nt = 0; nt < 8; ++nt) {
#pragma unroll
        for (int i = 0; i < 8; ++i) {
            int row = grp ? (i + 8) : i;
            unsafeAtomicAdd(
                &out[((size_t)tile * 16 + row) * D_MOD + wv * 128 + nt * 16 +
                     hl],
                tot[nt][i]);
        }
    }
}

// ---------------------------------------------------------------------------
// Launch
// ---------------------------------------------------------------------------
extern "C" void kernel_launch(void* const* d_in, const int* in_sizes, int n_in,
                              void* d_out, int out_size, void* d_ws,
                              size_t ws_size, hipStream_t stream) {
    const float* x  = (const float*)d_in[0];
    const float* Wg = (const float*)d_in[1];
    const float* w1 = (const float*)d_in[2];
    const float* w3 = (const float*)d_in[3];
    const float* w2 = (const float*)d_in[4];
    float* out = (float*)d_out;

    const size_t nX  = (size_t)T_TOK * D_MOD;
    const size_t nW  = (size_t)N_EXP * H_FFN * D_MOD;   // same for w1/w3/w2
    char* ws = (char*)d_ws;
    unsigned short* xb  = (unsigned short*)ws;
    unsigned short* w1b = (unsigned short*)(ws + nX * 2);
    unsigned short* w3b = (unsigned short*)(ws + nX * 2 + nW * 2);
    unsigned short* w2b = (unsigned short*)(ws + nX * 2 + nW * 4);
    float* comb         = (float*)(ws + nX * 2 + nW * 6);

    cvt_bf16<<<512, 256, 0, stream>>>(x, xb, (long)nX);
    cvt_bf16<<<4096, 256, 0, stream>>>(w1, w1b, (long)nW);
    cvt_bf16<<<4096, 256, 0, stream>>>(w3, w3b, (long)nW);
    cvt_bf16<<<4096, 256, 0, stream>>>(w2, w2b, (long)nW);

    zero_f32<<<512, 256, 0, stream>>>(out, (long)T_TOK * D_MOD);

    gate_topk<<<T_TOK / 256, 256, 0, stream>>>(x, Wg, comb);

    moe_main<<<dim3(T_TOK / 16, 2), 256, 0, stream>>>(xb, w1b, w3b, w2b, comb,
                                                      out);
}
